// SharedCausalSelfAttention_35407710388734
// MI455X (gfx1250) — matrix operand order repaired and checked by reading, not compiled, not run on hardware
//
#include <hip/hip_runtime.h>
#include <stdint.h>

#define TT 2048
#define EE 2048
#define HH 16
#define KVH 4
#define DD 128
#define WIN 1024

typedef __attribute__((ext_vector_type(16))) __bf16 v16bf;
typedef __attribute__((ext_vector_type(8)))  float  v8f;
typedef __attribute__((ext_vector_type(4)))  unsigned int v4u;
typedef __attribute__((ext_vector_type(8)))  unsigned int v8u;

struct Frag { union { v16bf v; uint4 u[2]; }; };

__device__ inline unsigned short f2bf(float x) {
  union { float f; unsigned int i; } c; c.f = x;
  unsigned int u = c.i;
  u += 0x7FFFu + ((u >> 16) & 1u);          // round-to-nearest-even
  return (unsigned short)(u >> 16);
}

// A-operand (16xK row-major source): lane(m=ln) holds 16 contiguous K at 16*half
__device__ inline void load_a_frag(Frag& f, const unsigned short* p, int half) {
  const uint4* q = reinterpret_cast<const uint4*>(p + half * 16);
  f.u[0] = q[0];
  f.u[1] = q[1];
}
// B-operand (N x K row-major source, i.e. B^T): lane(n=ln) holds 16 contiguous K at 16*half
__device__ inline void load_b_frag(Frag& f, const unsigned short* p, int half) {
  const uint4* q = reinterpret_cast<const uint4*>(p + half * 16);
  f.u[0] = q[0];
  f.u[1] = q[1];
}

// ---- Tensor Data Mover: stage a rows x 32 bf16 tile (row-major, row stride K elems)
// into LDS with 16B pad after each 64B row (LDS row stride = 80B). ISA 08 sect 8.3/8.4.
__device__ inline void tdm_load_2d(unsigned int lds_byte_off, const void* gsrc,
                                   unsigned int rows, unsigned int kelems) {
  unsigned long long ga = (unsigned long long)gsrc;
  v4u g0;
  g0[0] = 1u;                                    // count=1, user mode, no gather
  g0[1] = lds_byte_off;                          // lds_addr (bytes)
  g0[2] = (unsigned int)ga;                      // global_addr[31:0]
  g0[3] = (unsigned int)(ga >> 32) | (2u << 30); // global_addr[56:32] | type=2 (image)
  v8u g1;
  g1[0] = (1u << 16)        // data_size = 2 bytes
        | (1u << 20)        // pad_enable
        | (3u << 22)        // pad_interval: 16 DWORDs (one 64B row)
        | (3u << 25);       // pad_amount: 4 DWORDs (16B)
  g1[1] = (kelems & 0xFFFFu) << 16;                       // tensor_dim0[15:0]
  g1[2] = (kelems >> 16) | ((rows & 0xFFFFu) << 16);      // tensor_dim0[31:16] | tensor_dim1[15:0]
  g1[3] = (rows >> 16) | (32u << 16);                     // tensor_dim1[31:16] | tile_dim0=32
  g1[4] = rows & 0xFFFFu;                                 // tile_dim1 = rows, tile_dim2 = 0
  g1[5] = kelems;                                         // tensor_dim0_stride[31:0]
  g1[6] = 0u;                                             // stride[47:32] | t1_stride[15:0]
  g1[7] = 0u;
  asm volatile("tensor_load_to_lds %0, %1" :: "s"(g0), "s"(g1) : "memory");
}

// ---------------- prep kernels ----------------

__global__ __launch_bounds__(256) void k_fuse_mix(const float* __restrict__ W,
    const float* __restrict__ mix, unsigned short* __restrict__ out,
    int nh, int rowlen) {
  int idx = blockIdx.x * 256 + threadIdx.x;
  if (idx >= nh * rowlen) return;
  int h = idx / rowlen, j = idx - h * rowlen;
  float s = 0.f;
  for (int h2 = 0; h2 < nh; ++h2)
    s += mix[h * nh + h2] * W[(size_t)h2 * rowlen + j];
  out[idx] = f2bf(s);
}

__global__ __launch_bounds__(256) void k_f32_to_bf16(const float* __restrict__ src,
    unsigned short* __restrict__ dst, int n) {
  int i = blockIdx.x * 256 + threadIdx.x;
  if (i < n) dst[i] = f2bf(src[i]);
}

// ---- bf16 WMMA GEMM, TDM-staged double-buffered LDS: C[M][N] = A[M][K]*B[N][K]^T ----
// Block: 256 thr = 8 waves -> 128x256 C tile (64x64 per wave). LDS: 2x(128+256)x80B = 60KB.
__global__ __launch_bounds__(256) void k_gemm_tdm(
    const unsigned short* __restrict__ A, const unsigned short* __restrict__ B,
    float* __restrict__ C, int M, int N, int K) {
  __shared__ __align__(16) char smem[61440];
  unsigned int smbase = (unsigned int)(unsigned long long)&smem[0];
  int lane = threadIdx.x & 31, wave = threadIdx.x >> 5;
  int half = lane >> 4, ln = lane & 15;
  int mblk = blockIdx.y * 128, nblk = blockIdx.x * 256;
  int m0 = (wave >> 2) * 64, n0 = (wave & 3) * 64;
  bool issuer = (threadIdx.x < 32);

  if (issuer) {  // prologue: stage buffer 0
    tdm_load_2d(smbase + 0u,      A + (size_t)mblk * K, 128, (unsigned)K);
    tdm_load_2d(smbase + 20480u,  B + (size_t)nblk * K, 256, (unsigned)K);
  }
  const v8f z8 = {0.f,0.f,0.f,0.f,0.f,0.f,0.f,0.f};
  v8f c[4][4];
#pragma unroll
  for (int i = 0; i < 4; i++)
#pragma unroll
    for (int j = 0; j < 4; j++) c[i][j] = z8;

  int buf = 0;
  for (int kk = 0; kk < K; kk += 32, buf ^= 1) {
    if (issuer) {
      if (kk + 32 < K) {  // stage next buffer, then wait for previous stage (2 oldest)
        tdm_load_2d(smbase + (unsigned)(buf ^ 1) * 10240u,
                    A + (size_t)mblk * K + kk + 32, 128, (unsigned)K);
        tdm_load_2d(smbase + 20480u + (unsigned)(buf ^ 1) * 20480u,
                    B + (size_t)nblk * K + kk + 32, 256, (unsigned)K);
        __builtin_amdgcn_s_wait_tensorcnt(2);
      } else {
        __builtin_amdgcn_s_wait_tensorcnt(0);
      }
    }
    __syncthreads();   // tile in LDS visible to all waves
    const unsigned short* tA = (const unsigned short*)(smem + buf * 10240);
    const unsigned short* tB = (const unsigned short*)(smem + 20480 + buf * 20480);
    Frag a[4], b[4];
#pragma unroll
    for (int i = 0; i < 4; i++)
      load_a_frag(a[i], tA + (m0 + i * 16 + ln) * 40, half);   // 80B LDS row stride
#pragma unroll
    for (int j = 0; j < 4; j++)
      load_b_frag(b[j], tB + (n0 + j * 16 + ln) * 40, half);
#pragma unroll
    for (int i = 0; i < 4; i++)
#pragma unroll
      for (int j = 0; j < 4; j++)
        c[i][j] = __builtin_amdgcn_wmma_f32_16x16x32_bf16(
            false, a[i].v, false, b[j].v, (short)0, c[i][j], false, false);
    __syncthreads();   // all reads done before next TDM overwrites this buffer
  }
#pragma unroll
  for (int i = 0; i < 4; i++)
#pragma unroll
    for (int j = 0; j < 4; j++)
#pragma unroll
      for (int r = 0; r < 8; r++)
        C[(size_t)(mblk + m0 + i * 16 + r + 8 * half) * N + nblk + n0 + j * 16 + ln]
            = c[i][j][r];
}

// ---------------- RoPE + RMSNorm: one wave per (t, head) ----------------
__global__ __launch_bounds__(256) void k_rope_rms(const float* __restrict__ qf,
    const float* __restrict__ cosb, const float* __restrict__ sinb,
    unsigned short* __restrict__ qb, int nheads) {
  int lane = threadIdx.x & 31, wave = threadIdx.x >> 5;
  int widx = blockIdx.x * 8 + wave;
  int t = widx / nheads, hh = widx - t * nheads;
  const float* row = qf + (size_t)(t * nheads + hh) * DD;
  int d2 = lane * 2;
  float2 x1 = *reinterpret_cast<const float2*>(row + d2);
  float2 x2 = *reinterpret_cast<const float2*>(row + 64 + d2);
  float2 cc = *reinterpret_cast<const float2*>(cosb + t * 64 + d2);
  float2 ss = *reinterpret_cast<const float2*>(sinb + t * 64 + d2);
  float o1x =  x1.x * cc.x + x2.x * ss.x, o1y =  x1.y * cc.y + x2.y * ss.y;
  float o2x = -x1.x * ss.x + x2.x * cc.x, o2y = -x1.y * ss.y + x2.y * cc.y;
  float sq = o1x * o1x + o1y * o1y + o2x * o2x + o2y * o2y;
#pragma unroll
  for (int m = 1; m < 32; m <<= 1) sq += __shfl_xor(sq, m, 32);
  float rinv = rsqrtf(sq * (1.f / 128.f) + 1e-6f);
  unsigned short* orow = qb + (size_t)(t * nheads + hh) * DD;
  unsigned int lo = (unsigned int)f2bf(o1x * rinv) | ((unsigned int)f2bf(o1y * rinv) << 16);
  unsigned int hi = (unsigned int)f2bf(o2x * rinv) | ((unsigned int)f2bf(o2y * rinv) << 16);
  *reinterpret_cast<unsigned int*>(orow + d2) = lo;
  *reinterpret_cast<unsigned int*>(orow + 64 + d2) = hi;
}

// ---------------- V gate + transpose to [kv*D][T] bf16 ----------------
__global__ __launch_bounds__(256) void k_vprep(const float* __restrict__ vf,
    const float* __restrict__ ve, const float* __restrict__ ve_gate,
    unsigned short* __restrict__ vT) {
  int i = blockIdx.x * 256 + threadIdx.x;       // i < T * KVH * DD
  int c = i / TT, t = i - c * TT;               // t fastest -> coalesced writes
  float g = 2.f / (1.f + __expf(-ve_gate[c >> 7]));
  size_t src = (size_t)t * (KVH * DD) + c;
  vT[(size_t)c * TT + t] = f2bf(vf[src] + g * ve[src]);
}

// ---------------- flash attention: one wave per (16-query tile, head) ----------------
__global__ __launch_bounds__(128) void k_attn(
    const unsigned short* __restrict__ qb, const unsigned short* __restrict__ kb,
    const unsigned short* __restrict__ vT, const float* __restrict__ attn_gate,
    unsigned short* __restrict__ yb) {
  __shared__ unsigned short plds[4][16][40];    // padded: 80B row stride, conflict-free
  int lane = threadIdx.x & 31, wslot = threadIdx.x >> 5;
  int half = lane >> 4, ln = lane & 15;
  int w = blockIdx.x * 4 + wslot;
  int h = w & (HH - 1);
  int q0 = (w >> 4) * 16;
  int g = h >> 2;                               // GQA group

  Frag qa[4];
  {
    const unsigned short* qrow = qb + ((size_t)(q0 + ln) * HH + h) * DD;
#pragma unroll
    for (int c = 0; c < 4; c++) load_a_frag(qa[c], qrow + c * 32, half);
  }
  const v8f z8 = {0.f,0.f,0.f,0.f,0.f,0.f,0.f,0.f};
  v8f o[8];
#pragma unroll
  for (int t = 0; t < 8; t++) o[t] = z8;
  float mrow[8], lsum[8];
#pragma unroll
  for (int r = 0; r < 8; r++) { mrow[r] = -1e30f; lsum[r] = 0.f; }
  const float kscale = 0.088388347648318447f * 1.4426950408889634f; // 1/sqrt(128)*log2(e)

  int kk0 = (q0 >= WIN) ? ((q0 - WIN) & ~31) : 0;
  for (int kk = kk0; kk < q0 + 16; kk += 32) {
    v8f s[2]; s[0] = z8; s[1] = z8;
#pragma unroll
    for (int n = 0; n < 2; n++) {
      const unsigned short* krow = kb + ((size_t)(kk + n * 16 + ln) * KVH + g) * DD;
#pragma unroll
      for (int c = 0; c < 4; c++) {
        Frag kf; load_b_frag(kf, krow + c * 32, half);
        s[n] = __builtin_amdgcn_wmma_f32_16x16x32_bf16(
            false, qa[c].v, false, kf.v, (short)0, s[n], false, false);
      }
    }
    float sc[8];
#pragma unroll
    for (int r = 0; r < 8; r++) {
      int row = q0 + r + 8 * half;
#pragma unroll
      for (int n = 0; n < 2; n++) {
        int col = kk + n * 16 + ln;
        if (!((col <= row) && (row - col <= WIN))) s[n][r] = -1e30f;
      }
      float mx = fmaxf(s[0][r], s[1][r]);
#pragma unroll
      for (int msk = 1; msk < 16; msk <<= 1) mx = fmaxf(mx, __shfl_xor(mx, msk, 32));
      float mnew = fmaxf(mrow[r], mx);
      sc[r] = exp2f((mrow[r] - mnew) * kscale);
      mrow[r] = mnew;
      float p0 = exp2f((s[0][r] - mnew) * kscale);
      float p1 = exp2f((s[1][r] - mnew) * kscale);
      s[0][r] = p0; s[1][r] = p1;
      float rs = p0 + p1;
#pragma unroll
      for (int msk = 1; msk < 16; msk <<= 1) rs += __shfl_xor(rs, msk, 32);
      lsum[r] = lsum[r] * sc[r] + rs;
    }
#pragma unroll
    for (int t = 0; t < 8; t++)
#pragma unroll
      for (int r = 0; r < 8; r++) o[t][r] *= sc[r];

    // C-layout P -> LDS -> A-layout fragment (same-wave DS ops stay in order)
#pragma unroll
    for (int n = 0; n < 2; n++)
#pragma unroll
      for (int r = 0; r < 8; r++)
        plds[wslot][r + 8 * half][n * 16 + ln] = f2bf(s[n][r]);
    Frag pf;
    {
      const unsigned short* prow = &plds[wslot][ln][0];
      const uint4* pq = reinterpret_cast<const uint4*>(prow + half * 16);
      pf.u[0] = pq[0];
      pf.u[1] = pq[1];
    }
#pragma unroll
    for (int t = 0; t < 8; t++) {
      Frag vf2;
      load_b_frag(vf2, vT + ((size_t)g * DD + t * 16 + ln) * TT + kk, half);
      o[t] = __builtin_amdgcn_wmma_f32_16x16x32_bf16(
          false, pf.v, false, vf2.v, (short)0, o[t], false, false);
    }
  }
  float gate = 1.f / (1.f + __expf(-attn_gate[h]));
  float inv[8];
#pragma unroll
  for (int r = 0; r < 8; r++) inv[r] = gate / lsum[r];
#pragma unroll
  for (int t = 0; t < 8; t++)
#pragma unroll
    for (int r = 0; r < 8; r++)
      yb[(size_t)(q0 + r + 8 * half) * EE + h * DD + t * 16 + ln] = f2bf(o[t][r] * inv[r]);
}

// ---------------- host ----------------
extern "C" void kernel_launch(void* const* d_in, const int* in_sizes, int n_in,
                              void* d_out, int out_size, void* d_ws, size_t ws_size,
                              hipStream_t stream) {
  (void)in_sizes; (void)n_in; (void)out_size; (void)ws_size;
  const float* x        = (const float*)d_in[0];
  const float* ve       = (const float*)d_in[1];
  const float* cosb     = (const float*)d_in[2];
  const float* sinb     = (const float*)d_in[3];
  const float* ve_gate  = (const float*)d_in[4];
  const float* attng    = (const float*)d_in[5];
  const float* Wq       = (const float*)d_in[6];
  const float* Wk       = (const float*)d_in[7];
  const float* Wv       = (const float*)d_in[8];
  const float* Wo       = (const float*)d_in[9];
  const float* q_mix    = (const float*)d_in[10];
  const float* k_mix    = (const float*)d_in[11];
  const float* v_mix    = (const float*)d_in[12];

  char* ws = (char*)d_ws;
  size_t off = 0;
  auto alloc = [&](size_t bytes) -> char* {
    char* p = ws + off;
    off = (off + bytes + 255) & ~(size_t)255;
    return p;
  };
  unsigned short* Wqb = (unsigned short*)alloc((size_t)EE * EE * 2);
  unsigned short* Wkb = (unsigned short*)alloc((size_t)KVH * DD * EE * 2);
  unsigned short* Wvb = (unsigned short*)alloc((size_t)KVH * DD * EE * 2);
  unsigned short* Wob = (unsigned short*)alloc((size_t)EE * EE * 2);
  unsigned short* xb  = (unsigned short*)alloc((size_t)TT * EE * 2);
  float*          qf  = (float*)alloc((size_t)TT * EE * 4);
  float*          kf  = (float*)alloc((size_t)TT * KVH * DD * 4);
  float*          vf  = (float*)alloc((size_t)TT * KVH * DD * 4);
  unsigned short* qbf = (unsigned short*)alloc((size_t)TT * EE * 2);
  unsigned short* kbf = (unsigned short*)alloc((size_t)TT * KVH * DD * 2);
  unsigned short* vTb = (unsigned short*)alloc((size_t)KVH * DD * TT * 2);
  unsigned short* yb  = (unsigned short*)alloc((size_t)TT * EE * 2);

  const int RL = DD * EE;  // reshape row length for mix fusion
  // 1. weight fusion / casts
  k_fuse_mix<<<(HH * RL + 255) / 256, 256, 0, stream>>>(Wq, q_mix, Wqb, HH, RL);
  k_fuse_mix<<<(KVH * RL + 255) / 256, 256, 0, stream>>>(Wk, k_mix, Wkb, KVH, RL);
  k_fuse_mix<<<(KVH * RL + 255) / 256, 256, 0, stream>>>(Wv, v_mix, Wvb, KVH, RL);
  k_f32_to_bf16<<<(EE * EE + 255) / 256, 256, 0, stream>>>(Wo, Wob, EE * EE);
  k_f32_to_bf16<<<(TT * EE + 255) / 256, 256, 0, stream>>>(x, xb, TT * EE);

  // 2. projections (TDM-staged WMMA)
  k_gemm_tdm<<<dim3(EE / 256, TT / 128), 256, 0, stream>>>(xb, Wqb, qf, TT, EE, EE);
  k_gemm_tdm<<<dim3((KVH * DD) / 256, TT / 128), 256, 0, stream>>>(xb, Wkb, kf, TT, KVH * DD, EE);
  k_gemm_tdm<<<dim3((KVH * DD) / 256, TT / 128), 256, 0, stream>>>(xb, Wvb, vf, TT, KVH * DD, EE);

  // 3. RoPE+RMSNorm for q,k ; gate+transpose for v
  k_rope_rms<<<(TT * HH) / 8, 256, 0, stream>>>(qf, cosb, sinb, qbf, HH);
  k_rope_rms<<<(TT * KVH) / 8, 256, 0, stream>>>(kf, cosb, sinb, kbf, KVH);
  k_vprep<<<(TT * KVH * DD) / 256, 256, 0, stream>>>(vf, ve, ve_gate, vTb);

  // 4. windowed-causal GQA flash attention (WMMA)
  k_attn<<<(TT / 16) * HH / 4, 128, 0, stream>>>(qbf, kbf, vTb, attng, yb);

  // 5. output projection (TDM-staged WMMA, fp32 out)
  k_gemm_tdm<<<dim3(EE / 256, TT / 128), 256, 0, stream>>>(yb, Wob, (float*)d_out, TT, EE, EE);
}
